// LongformerModelText_76012331204994
// MI455X (gfx1250) — matrix operand order repaired and checked by reading, not compiled, not run on hardware
//
#include <hip/hip_runtime.h>
#include <math.h>

typedef __bf16 bf16_t;
typedef __attribute__((ext_vector_type(16))) __bf16 v16bf;
typedef __attribute__((ext_vector_type(8)))  __bf16 v8bf;
typedef __attribute__((ext_vector_type(8)))  float  v8f;
typedef __attribute__((ext_vector_type(4)))  int    v4i_t;

#define SB   2
#define SS   4096
#define SD   768
#define SH   12
#define SDH  64
#define SL   12
#define SDFF 3072
#define SW   32
#define SC   (SS / SW)
#define NTOK (SB * SS)
#define SNC  16

#if defined(__has_builtin)
#if __has_builtin(__builtin_amdgcn_global_load_async_to_lds_b128)
#define HAVE_ASYNC_LDS 1
#endif
#endif

typedef __attribute__((address_space(1))) v4i_t* gv4p;   // global int4*
typedef __attribute__((address_space(3))) v4i_t* lv4p;   // LDS int4*

__device__ __forceinline__ void stage_b128(const bf16_t* gsrc, bf16_t* ldst)
{
#ifdef HAVE_ASYNC_LDS
    __builtin_amdgcn_global_load_async_to_lds_b128((gv4p)gsrc, (lv4p)ldst, 0, 0);
#else
    *(v8bf*)ldst = *(const v8bf*)gsrc;
#endif
}

__device__ __forceinline__ void stage_wait()
{
#ifdef HAVE_ASYNC_LDS
#if __has_builtin(__builtin_amdgcn_s_wait_asynccnt)
    __builtin_amdgcn_s_wait_asynccnt(0);
#else
    asm volatile("s_wait_asynccnt 0x0" ::: "memory");
#endif
#endif
}

// ---------------------------------------------------------------------------
// Embedding + LayerNorm:  h = LN(emb[x] + pos_emb[s+1] + tok_emb)
// ---------------------------------------------------------------------------
__global__ __launch_bounds__(256) void k_embed(
    const int* __restrict__ x, const float* __restrict__ emb,
    const float* __restrict__ pos, const float* __restrict__ tok,
    const float* __restrict__ gam, const float* __restrict__ bet,
    float* __restrict__ h, bf16_t* __restrict__ hb)
{
    __shared__ float red[256];
    __shared__ float vals[SD];
    const int t = blockIdx.x;
    const int s = t % SS;
    const int id = x[t];

    float lsum = 0.f;
    for (int d = threadIdx.x; d < SD; d += 256) {
        float v = emb[(size_t)id * SD + d] + pos[(size_t)(s + 1) * SD + d] + tok[d];
        vals[d] = v;
        lsum += v;
    }
    red[threadIdx.x] = lsum;
    __syncthreads();
    for (int o = 128; o > 0; o >>= 1) {
        if (threadIdx.x < o) red[threadIdx.x] += red[threadIdx.x + o];
        __syncthreads();
    }
    const float mean = red[0] * (1.0f / SD);
    __syncthreads();

    float ls2 = 0.f;
    for (int d = threadIdx.x; d < SD; d += 256) {
        float dv = vals[d] - mean;
        ls2 += dv * dv;
    }
    red[threadIdx.x] = ls2;
    __syncthreads();
    for (int o = 128; o > 0; o >>= 1) {
        if (threadIdx.x < o) red[threadIdx.x] += red[threadIdx.x + o];
        __syncthreads();
    }
    const float rstd = rsqrtf(red[0] * (1.0f / SD) + 1e-12f);

    for (int d = threadIdx.x; d < SD; d += 256) {
        float o = (vals[d] - mean) * rstd * gam[d] + bet[d];
        h[(size_t)t * SD + d]  = o;
        hb[(size_t)t * SD + d] = (bf16_t)o;
    }
}

// ---------------------------------------------------------------------------
// LayerNorm over residual-summed f32 input; writes f32 + bf16.
// ---------------------------------------------------------------------------
__global__ __launch_bounds__(256) void k_ln(
    const float* __restrict__ in, const float* __restrict__ gam,
    const float* __restrict__ bet, float* __restrict__ h, bf16_t* __restrict__ hb)
{
    __shared__ float red[256];
    const int t = blockIdx.x;
    const float* row = in + (size_t)t * SD;

    float lsum = 0.f;
    for (int d = threadIdx.x; d < SD; d += 256) lsum += row[d];
    red[threadIdx.x] = lsum;
    __syncthreads();
    for (int o = 128; o > 0; o >>= 1) {
        if (threadIdx.x < o) red[threadIdx.x] += red[threadIdx.x + o];
        __syncthreads();
    }
    const float mean = red[0] * (1.0f / SD);
    __syncthreads();

    float ls2 = 0.f;
    for (int d = threadIdx.x; d < SD; d += 256) {
        float dv = row[d] - mean;
        ls2 += dv * dv;
    }
    red[threadIdx.x] = ls2;
    __syncthreads();
    for (int o = 128; o > 0; o >>= 1) {
        if (threadIdx.x < o) red[threadIdx.x] += red[threadIdx.x + o];
        __syncthreads();
    }
    const float rstd = rsqrtf(red[0] * (1.0f / SD) + 1e-12f);

    for (int d = threadIdx.x; d < SD; d += 256) {
        float o = (row[d] - mean) * rstd * gam[d] + bet[d];
        h[(size_t)t * SD + d]  = o;
        hb[(size_t)t * SD + d] = (bf16_t)o;
    }
}

// ---------------------------------------------------------------------------
// Weight convert f32 [K,N] -> bf16 transposed [N,K].
// ---------------------------------------------------------------------------
__global__ void k_cvtwT(const float* __restrict__ Wsrc, bf16_t* __restrict__ Wt,
                        int K, int N)
{
    long idx = (long)blockIdx.x * 256 + threadIdx.x;
    if (idx < (long)K * N) {
        int k = (int)(idx / N);
        int n = (int)(idx % N);
        Wt[(long)n * K + k] = (bf16_t)Wsrc[idx];
    }
}

// ---------------------------------------------------------------------------
// WMMA bf16 GEMM:  C[M,N] = A[M,K] * Bt[N,K]^T  (+bias)(+resid)(gelu->bf16)
// Block tile 128x128, 8 waves (4m x 2n), wave tile 32x64 = 2x4 wmma 16x16x32.
// EPI: 0 = f32 store (+bias); 1 = f32 store (+bias+resid);
//      2 = bf16 store (+bias, exact GELU)
// ---------------------------------------------------------------------------
template <int EPI>
__global__ __launch_bounds__(256) void k_gemm(
    const bf16_t* __restrict__ A, const bf16_t* __restrict__ Bt,
    const float* __restrict__ bias, const float* __restrict__ resid,
    float* __restrict__ Cf, bf16_t* __restrict__ Cb,
    int M, int N, int K)
{
    __shared__ bf16_t As[128][40];   // 32 K-halves + 8 pad (80B row stride)
    __shared__ bf16_t Bs[128][40];

    const int tid   = threadIdx.x;
    const int lane  = tid & 31;
    const int wave  = tid >> 5;
    const int wm    = wave >> 1;      // 0..3  (32 rows each)
    const int wn    = wave & 1;       // 0..1  (64 cols each)
    const int l16   = lane & 15;
    const int lhalf = lane >> 4;      // 0/1
    const int bm0   = blockIdx.y * 128;
    const int bn0   = blockIdx.x * 128;

    v8f acc[2][4] = {};

    const int akb = lhalf * 8;        // A: lanes 16-31 start at K=8  (ISA 7.12.2)
    const int bkb = lhalf * 8;        // B: lanes 16-31 start at K=8

    // staging decomposition: 512 chunks of 8 halves per 128x32 tile
    const int sr = tid >> 1;                 // 0..127 row
    const int sc0 = (tid & 1) << 4;          // 0 / 16 (two 8-half chunks each)

    for (int k0 = 0; k0 < K; k0 += 32) {
        stage_b128(A  + (size_t)(bm0 + sr) * K + k0 + sc0,     &As[sr][sc0]);
        stage_b128(A  + (size_t)(bm0 + sr) * K + k0 + sc0 + 8, &As[sr][sc0 + 8]);
        stage_b128(Bt + (size_t)(bn0 + sr) * K + k0 + sc0,     &Bs[sr][sc0]);
        stage_b128(Bt + (size_t)(bn0 + sr) * K + k0 + sc0 + 8, &Bs[sr][sc0 + 8]);
        stage_wait();
        __syncthreads();

        v16bf afrag[2], bfrag[4];
        for (int mi = 0; mi < 2; ++mi) {
            int row = wm * 32 + mi * 16 + l16;
            v8bf lo = *(const v8bf*)&As[row][akb];
            v8bf hi = *(const v8bf*)&As[row][16 + akb];
            afrag[mi] = __builtin_shufflevector(lo, hi,
                0,1,2,3,4,5,6,7,8,9,10,11,12,13,14,15);
        }
        for (int ni = 0; ni < 4; ++ni) {
            int rowb = wn * 64 + ni * 16 + l16;
            v8bf lo = *(const v8bf*)&Bs[rowb][bkb];
            v8bf hi = *(const v8bf*)&Bs[rowb][16 + bkb];
            bfrag[ni] = __builtin_shufflevector(lo, hi,
                0,1,2,3,4,5,6,7,8,9,10,11,12,13,14,15);
        }
        for (int mi = 0; mi < 2; ++mi)
            for (int ni = 0; ni < 4; ++ni)
                acc[mi][ni] = __builtin_amdgcn_wmma_f32_16x16x32_bf16(
                    false, afrag[mi], false, bfrag[ni],
                    (short)0, acc[mi][ni], false, false);
        __syncthreads();
    }

    // Epilogue. C/D layout: VGPR r, lanes 0-15 -> M=r, lanes 16-31 -> M=8+r; N=l16.
    for (int mi = 0; mi < 2; ++mi)
        for (int ni = 0; ni < 4; ++ni) {
            const int n = bn0 + wn * 64 + ni * 16 + l16;
            const int mbase = bm0 + wm * 32 + mi * 16 + lhalf * 8;
            for (int r = 0; r < 8; ++r) {
                const size_t off = (size_t)(mbase + r) * N + n;
                float v = acc[mi][ni][r] + bias[n];
                if (EPI == 0) {
                    Cf[off] = v;
                } else if (EPI == 1) {
                    Cf[off] = v + resid[off];
                } else {
                    Cb[off] = (bf16_t)(0.5f * v *
                                       (1.0f + erff(v * 0.70710678118654752f)));
                }
            }
        }
}

// ---------------------------------------------------------------------------
// Sliding-window band attention. One block per (chunk c, head, batch).
// ---------------------------------------------------------------------------
__global__ __launch_bounds__(256) void k_attn(
    const float* __restrict__ qf, const float* __restrict__ kf,
    const float* __restrict__ vf, bf16_t* __restrict__ ab)
{
    __shared__ float qs[SW][SDH];        //  8 KB
    __shared__ float kv[3 * SW][SDH];    // 24 KB (k during scores, v after)
    __shared__ float sc[SW][3 * SW];     // 12 KB

    const int c  = blockIdx.x;
    const int hh = blockIdx.y;
    const int b  = blockIdx.z;
    const size_t baseTok = (size_t)b * SS;
    const int col0 = hh * SDH;

    for (int idx = threadIdx.x; idx < SW * SDH; idx += 256) {
        int p = idx >> 6, d = idx & 63;
        qs[p][d] = qf[(baseTok + c * SW + p) * SD + col0 + d] * 0.125f; // 1/sqrt(64)
    }
    for (int idx = threadIdx.x; idx < 3 * SW * SDH; idx += 256) {
        int g = idx >> 6, d = idx & 63;
        int j = (c - 1) * SW + g;
        kv[g][d] = (j >= 0 && j < SS) ? kf[(baseTok + j) * SD + col0 + d] : 0.f;
    }
    __syncthreads();

    for (int idx = threadIdx.x; idx < SW * 3 * SW; idx += 256) {
        int p = idx / (3 * SW), g = idx % (3 * SW);
        int j = (c - 1) * SW + g;
        float vout = -INFINITY;
        if (g >= p && g <= p + 2 * SW && j >= 0 && j < SS) {
            float dot = 0.f;
            for (int d = 0; d < SDH; ++d) dot += qs[p][d] * kv[g][d];
            vout = dot;
        }
        sc[p][g] = vout;
    }
    __syncthreads();

    for (int idx = threadIdx.x; idx < 3 * SW * SDH; idx += 256) {
        int g = idx >> 6, d = idx & 63;
        int j = (c - 1) * SW + g;
        kv[g][d] = (j >= 0 && j < SS) ? vf[(baseTok + j) * SD + col0 + d] : 0.f;
    }
    if (threadIdx.x < SW) {
        int p = threadIdx.x;
        float mx = -INFINITY;
        for (int g = 0; g < 3 * SW; ++g) mx = fmaxf(mx, sc[p][g]);
        float sum = 0.f;
        for (int g = 0; g < 3 * SW; ++g) {
            float e = expf(sc[p][g] - mx);   // masked -> 0
            sc[p][g] = e;
            sum += e;
        }
        float inv = 1.f / sum;
        for (int g = 0; g < 3 * SW; ++g) sc[p][g] *= inv;
    }
    __syncthreads();

    for (int idx = threadIdx.x; idx < SW * SDH; idx += 256) {
        int p = idx >> 6, d = idx & 63;
        float acc = 0.f;
        for (int g = 0; g < 3 * SW; ++g) acc += sc[p][g] * kv[g][d];
        ab[(baseTok + c * SW + p) * SD + col0 + d] = (bf16_t)acc;
    }
}

// ---------------------------------------------------------------------------
// Mean over sequence, then tiny classification head.
// ---------------------------------------------------------------------------
__global__ void k_mean(const float* __restrict__ h, float* __restrict__ mean)
{
    int d = blockIdx.x * 256 + threadIdx.x;
    if (d < SB * SD) {
        int b = d / SD, dd = d % SD;
        const float* p = h + (size_t)b * SS * SD + dd;
        float s = 0.f;
        for (int i = 0; i < SS; ++i) s += p[(size_t)i * SD];
        mean[d] = s * (1.0f / SS);
    }
}

__global__ void k_head(const float* __restrict__ mean, const float* __restrict__ fcw,
                       const float* __restrict__ fcb, float* __restrict__ out)
{
    int t = threadIdx.x;
    if (t < SB * SNC) {
        int b = t / SNC, n = t % SNC;
        float s = fcb[n];
        for (int d = 0; d < SD; ++d) s += mean[b * SD + d] * fcw[d * SNC + n];
        out[t] = s;
    }
}

// ---------------------------------------------------------------------------
// Host driver
// ---------------------------------------------------------------------------
extern "C" void kernel_launch(void* const* d_in, const int* in_sizes, int n_in,
                              void* d_out, int out_size, void* d_ws, size_t ws_size,
                              hipStream_t stream)
{
    (void)in_sizes; (void)n_in; (void)out_size; (void)ws_size;

    const int*   x      = (const int*)  d_in[0];
    const float* emb    = (const float*)d_in[1];
    const float* pose   = (const float*)d_in[2];
    const float* toke   = (const float*)d_in[3];
    const float* elns   = (const float*)d_in[4];
    const float* elnb   = (const float*)d_in[5];
    const float* Wq     = (const float*)d_in[6];
    const float* bq     = (const float*)d_in[7];
    const float* Wk     = (const float*)d_in[8];
    const float* bk     = (const float*)d_in[9];
    const float* Wv     = (const float*)d_in[10];
    const float* bv     = (const float*)d_in[11];
    const float* Wo     = (const float*)d_in[12];
    const float* bo     = (const float*)d_in[13];
    const float* ln1s   = (const float*)d_in[14];
    const float* ln1b   = (const float*)d_in[15];
    const float* W1     = (const float*)d_in[16];
    const float* b1     = (const float*)d_in[17];
    const float* W2     = (const float*)d_in[18];
    const float* b2     = (const float*)d_in[19];
    const float* ln2s   = (const float*)d_in[20];
    const float* ln2b   = (const float*)d_in[21];
    const float* fcw    = (const float*)d_in[22];
    const float* fcb    = (const float*)d_in[23];
    float* out = (float*)d_out;

    char* ws = (char*)d_ws;
    auto carve = [&](size_t bytes) {
        char* p = ws;
        ws += (bytes + 255) & ~(size_t)255;
        return p;
    };
    const size_t NE = (size_t)NTOK * SD;
    float*  h    = (float*) carve(NE * 4);
    bf16_t* hb   = (bf16_t*)carve(NE * 2);
    float*  qf   = (float*) carve(NE * 4);
    float*  kfb  = (float*) carve(NE * 4);
    float*  vfb  = (float*) carve(NE * 4);
    bf16_t* abf  = (bf16_t*)carve(NE * 2);
    float*  tmp  = (float*) carve(NE * 4);
    bf16_t* ffb  = (bf16_t*)carve((size_t)NTOK * SDFF * 2);
    bf16_t* wbuf = (bf16_t*)carve((size_t)SD * SDFF * 2);
    float*  mbuf = (float*) carve((size_t)SB * SD * 4);

    auto cvtw = [&](const float* src, int K, int N) {
        long elems = (long)K * N;
        k_cvtwT<<<(unsigned)((elems + 255) / 256), 256, 0, stream>>>(src, wbuf, K, N);
    };
    auto gemm = [&](const bf16_t* A, const float* bias, const float* resid,
                    float* Cf, bf16_t* Cb, int N, int K, int epi) {
        dim3 grid(N / 128, NTOK / 128);
        if (epi == 0)
            k_gemm<0><<<grid, 256, 0, stream>>>(A, wbuf, bias, resid, Cf, Cb, NTOK, N, K);
        else if (epi == 1)
            k_gemm<1><<<grid, 256, 0, stream>>>(A, wbuf, bias, resid, Cf, Cb, NTOK, N, K);
        else
            k_gemm<2><<<grid, 256, 0, stream>>>(A, wbuf, bias, resid, Cf, Cb, NTOK, N, K);
    };

    k_embed<<<NTOK, 256, 0, stream>>>(x, emb, pose, toke, elns, elnb, h, hb);

    for (int l = 0; l < SL; ++l) {
        const size_t oDD = (size_t)l * SD * SD;
        const size_t oDF = (size_t)l * SD * SDFF;

        cvtw(Wq + oDD, SD, SD);
        gemm(hb, bq + (size_t)l * SD, nullptr, qf, nullptr, SD, SD, 0);
        cvtw(Wk + oDD, SD, SD);
        gemm(hb, bk + (size_t)l * SD, nullptr, kfb, nullptr, SD, SD, 0);
        cvtw(Wv + oDD, SD, SD);
        gemm(hb, bv + (size_t)l * SD, nullptr, vfb, nullptr, SD, SD, 0);

        k_attn<<<dim3(SC, SH, SB), 256, 0, stream>>>(qf, kfb, vfb, abf);

        cvtw(Wo + oDD, SD, SD);
        gemm(abf, bo + (size_t)l * SD, h, tmp, nullptr, SD, SD, 1);
        k_ln<<<NTOK, 256, 0, stream>>>(tmp, ln1s + (size_t)l * SD,
                                       ln1b + (size_t)l * SD, h, hb);

        cvtw(W1 + oDF, SD, SDFF);
        gemm(hb, b1 + (size_t)l * SDFF, nullptr, nullptr, ffb, SDFF, SD, 2);
        cvtw(W2 + oDF, SDFF, SD);
        gemm(ffb, b2 + (size_t)l * SD, h, tmp, nullptr, SD, SDFF, 1);
        k_ln<<<NTOK, 256, 0, stream>>>(tmp, ln2s + (size_t)l * SD,
                                       ln2b + (size_t)l * SD, h, hb);
    }

    k_mean<<<(SB * SD + 255) / 256, 256, 0, stream>>>(h, mbuf);
    k_head<<<1, 32, 0, stream>>>(mbuf, fcw, fcb, out);
}
